// ChannelTail_2843268350071
// MI455X (gfx1250) — matrix operand order repaired and checked by reading, not compiled, not run on hardware
//
#include <hip/hip_runtime.h>

typedef __attribute__((ext_vector_type(8)))  float   v8f;
typedef __attribute__((ext_vector_type(16))) __bf16  v16bf;
typedef __attribute__((ext_vector_type(8)))  __bf16  v8bf;

// Problem constants
#define BB    8
#define CIN   512
#define CINT  256
#define HW    16384   // 128*128

// ---------------------------------------------------------------------------
// Kernel 1: softmax over energy rows [B*CINT rows of CINT] + bias_eff = attn.b_value
// ---------------------------------------------------------------------------
__global__ __launch_bounds__(256) void ct_softmax_bias(const float* __restrict__ E,
                                                       const float* __restrict__ b_value,
                                                       float* __restrict__ attn,
                                                       float* __restrict__ bias_eff) {
    __shared__ float s[256];
    const int row = blockIdx.x;            // b*256 + i
    const int t   = threadIdx.x;
    const float v = E[(size_t)row * CINT + t];

    s[t] = v; __syncthreads();
    for (int off = 128; off > 0; off >>= 1) {
        if (t < off) s[t] = fmaxf(s[t], s[t + off]);
        __syncthreads();
    }
    const float mx = s[0]; __syncthreads();

    const float e = __expf(v - mx);
    s[t] = e; __syncthreads();
    for (int off = 128; off > 0; off >>= 1) {
        if (t < off) s[t] += s[t + off];
        __syncthreads();
    }
    const float inv = 1.0f / s[0]; __syncthreads();

    const float a = e * inv;
    attn[(size_t)row * CINT + t] = a;

    s[t] = a * b_value[t]; __syncthreads();
    for (int off = 128; off > 0; off >>= 1) {
        if (t < off) s[t] += s[t + off];
        __syncthreads();
    }
    if (t == 0) bias_eff[row] = s[0];
}

// ---------------------------------------------------------------------------
// Kernel 2: A_eff[b] = attn[b] (256x256) @ w_value (256x512)  -> [B,256,512] f32
// ---------------------------------------------------------------------------
__global__ __launch_bounds__(256) void ct_aeff(const float* __restrict__ attn,
                                               const float* __restrict__ w_value,
                                               float* __restrict__ A_eff) {
    const int idx = blockIdx.x * 256 + threadIdx.x;   // b*(256*512) + i*512 + c
    const int c = idx & 511;
    const int i = (idx >> 9) & 255;
    const int b = idx >> 17;
    const float* ar = attn + ((size_t)b * CINT + i) * CINT;
    float acc = 0.0f;
    #pragma unroll 4
    for (int j = 0; j < CINT; ++j)
        acc = fmaf(ar[j], w_value[(size_t)j * CIN + c], acc);
    A_eff[(size_t)idx] = acc;
}

// ---------------------------------------------------------------------------
// Kernel 3: W_eff[b] = w_re (512x256) @ A_eff[b] (256x512) -> [B,512,512] bf16
// ---------------------------------------------------------------------------
__global__ __launch_bounds__(256) void ct_weff(const float* __restrict__ w_re,
                                               const float* __restrict__ A_eff,
                                               __bf16* __restrict__ Weff) {
    const int idx = blockIdx.x * 256 + threadIdx.x;   // b*(512*512) + m*512 + k
    const int k = idx & 511;
    const int m = (idx >> 9) & 511;
    const int b = idx >> 18;
    const float* wr = w_re + (size_t)m * CINT;
    const float* ac = A_eff + (size_t)b * CINT * CIN + k;
    float acc = 0.0f;
    #pragma unroll 4
    for (int c = 0; c < CINT; ++c)
        acc = fmaf(wr[c], ac[(size_t)c * CIN], acc);
    Weff[(size_t)idx] = (__bf16)acc;
}

// ---------------------------------------------------------------------------
// Kernel 4: bias2[b][m] = w_re[m,:] . bias_eff[b,:] + b_re[m]
// ---------------------------------------------------------------------------
__global__ __launch_bounds__(256) void ct_bias2(const float* __restrict__ w_re,
                                                const float* __restrict__ bias_eff,
                                                const float* __restrict__ b_re,
                                                float* __restrict__ bias2) {
    const int idx = blockIdx.x * 256 + threadIdx.x;   // b*512 + m
    const int m = idx & 511;
    const int b = idx >> 9;
    const float* wr = w_re + (size_t)m * CINT;
    const float* be = bias_eff + (size_t)b * CINT;
    float acc = b_re[m];
    #pragma unroll 4
    for (int c = 0; c < CINT; ++c)
        acc = fmaf(wr[c], be[c], acc);
    bias2[idx] = acc;
}

// ---------------------------------------------------------------------------
// Kernel 5: per batch, out = gamma*(W_eff @ x + bias2) + 2*x
// bf16 WMMA, f32 accumulate. Block: 512 threads = 16 waves (8 M-waves x 2
// N-waves); workgroup tile M=512 x N=64; per-wave tile 64x32 (8 accumulators).
// K loop step 32, double-buffered LDS x-tile (transposed [n][k] bf16),
// register-staged global x load pipelined one iteration ahead, and all 8
// A-fragment b128 loads per K-step issued as one clause ahead of the WMMAs.
// ---------------------------------------------------------------------------
__global__ __launch_bounds__(512) void ct_main_gemm(const float* __restrict__ X,
                                                    const __bf16* __restrict__ Weff,
                                                    const float* __restrict__ bias2,
                                                    const float* __restrict__ gammap,
                                                    float* __restrict__ Out) {
    constexpr int K     = CIN;     // 512 (reduction dim)
    constexpr int NT    = 64;      // N tile per workgroup
    constexpr int KC    = 32;      // K step
    constexpr int NITER = K / KC;  // 16
    // rows padded to 40 bf16 (80B): 16B alignment for b128 LDS loads + bank spread
    __shared__ __bf16 xs[2][NT][40];

    const int b    = blockIdx.y;
    const int n0   = blockIdx.x * NT;
    const int tid  = threadIdx.x;
    const int wave = tid >> 5;
    const int lane = tid & 31;
    const int mw   = wave >> 1;          // 0..7  -> M
    const int nw   = wave & 1;           // 0..1  -> N half
    const int sel  = lane >> 4;          // half-wave select
    const int lr   = lane & 15;
    const int m_base = mw * 64;

    const float g = gammap[0];

    const size_t xbase = (size_t)b * K * HW;
    const __bf16* wbase = Weff + (size_t)b * CIN * K;   // [m][k], stride K

    // loader coords: 512 threads, tile = 32 K x 64 N f32, one float4 each
    const int lk = tid >> 4;          // 0..31  (K within tile)
    const int ln = (tid & 15) * 4;    // 0..60  (N within tile)
    const float* xptr = X + xbase + (size_t)lk * HW + n0 + ln;

    v8f acc[4][2] = {};

    // preload first tile into registers
    float4 v = *reinterpret_cast<const float4*>(xptr);

    for (int it = 0; it < NITER; ++it) {
        const int buf = it & 1;
        const int k0  = it * KC;

        // issue ALL A-fragment loads for this K-step first (one clause of 8
        // global_load_b128); WMMAs below then retire under staggered waits.
        // Layout (M x K = 16x32 bf16): lane lr = row; lanes 0-15 hold
        // K {0..7,16..23}, lanes 16-31 hold K {8..15,24..31}.
        v8bf alo[4], ahi[4];
        #pragma unroll
        for (int ms = 0; ms < 4; ++ms) {
            const __bf16* ap = wbase + (size_t)(m_base + ms * 16 + lr) * K
                                     + k0 + sel * 8;
            alo[ms] = *reinterpret_cast<const v8bf*>(ap);
            ahi[ms] = *reinterpret_cast<const v8bf*>(ap + 16);
        }

        // commit staged x tile to LDS (bf16, transposed [n][k])
        xs[buf][ln + 0][lk] = (__bf16)v.x;
        xs[buf][ln + 1][lk] = (__bf16)v.y;
        xs[buf][ln + 2][lk] = (__bf16)v.z;
        xs[buf][ln + 3][lk] = (__bf16)v.w;
        __syncthreads();

        // issue next x tile load immediately; consumed only next iteration,
        // so the WMMA span below hides the memory latency
        if (it + 1 < NITER)
            v = *reinterpret_cast<const float4*>(xptr + (size_t)(it + 1) * KC * HW);
        if (it + 2 < NITER)   // global_prefetch_b8 two tiles ahead
            __builtin_prefetch(xptr + (size_t)(it + 2) * KC * HW, 0, 1);

        // B fragments (K x N = 32x16 bf16): lanes 0-15 hold K {0..7,16..23} of
        // column N=lane; lanes 16-31 hold K {8..15,24..31} of column N=lane-16.
        v16bf bfrag[2];
        #pragma unroll
        for (int ns = 0; ns < 2; ++ns) {
            const __bf16* p = &xs[buf][nw * 32 + ns * 16 + lr][sel * 8];
            v8bf lo = *reinterpret_cast<const v8bf*>(p);
            v8bf hi = *reinterpret_cast<const v8bf*>(p + 16);
            bfrag[ns] = __builtin_shufflevector(lo, hi,
                0,1,2,3,4,5,6,7,8,9,10,11,12,13,14,15);
        }

        #pragma unroll
        for (int ms = 0; ms < 4; ++ms) {
            v16bf afrag = __builtin_shufflevector(alo[ms], ahi[ms],
                0,1,2,3,4,5,6,7,8,9,10,11,12,13,14,15);
            #pragma unroll
            for (int ns = 0; ns < 2; ++ns) {
                acc[ms][ns] = __builtin_amdgcn_wmma_f32_16x16x32_bf16(
                    false, afrag, false, bfrag[ns],
                    (short)0, acc[ms][ns], false, false);
            }
        }
    }

    // Epilogue: C/D layout: VGPR r -> M = r + 8*sel ; N = lr
    #pragma unroll
    for (int ms = 0; ms < 4; ++ms) {
        #pragma unroll
        for (int ns = 0; ns < 2; ++ns) {
            #pragma unroll
            for (int r = 0; r < 8; ++r) {
                const int m = m_base + ms * 16 + sel * 8 + r;
                const int n = n0 + nw * 32 + ns * 16 + lr;
                const size_t oi = xbase + (size_t)m * HW + n;
                const float xv = X[oi];
                Out[oi] = g * (acc[ms][ns][r] + bias2[b * CIN + m]) + 2.0f * xv;
            }
        }
    }
}

// ---------------------------------------------------------------------------
// Launch
// ---------------------------------------------------------------------------
extern "C" void kernel_launch(void* const* d_in, const int* in_sizes, int n_in,
                              void* d_out, int out_size, void* d_ws, size_t ws_size,
                              hipStream_t stream) {
    const float* energy  = (const float*)d_in[0];   // [8,256,256]
    const float* x       = (const float*)d_in[1];   // [8,512,128,128]
    const float* w_value = (const float*)d_in[2];   // [256,512]
    const float* b_value = (const float*)d_in[3];   // [256]
    const float* w_re    = (const float*)d_in[4];   // [512,256]
    const float* b_re    = (const float*)d_in[5];   // [512]
    const float* gamma   = (const float*)d_in[6];   // [1]
    float* out = (float*)d_out;                     // [8,512,128,128]

    char* ws = (char*)d_ws;
    float*  attn     = (float*)(ws + 0);            // 8*256*256*4 = 2 MiB
    float*  A_eff    = (float*)(ws + 2097152);      // 8*256*512*4 = 4 MiB
    float*  bias_eff = (float*)(ws + 6291456);      // 8*256*4
    float*  bias2    = (float*)(ws + 6299648);      // 8*512*4
    __bf16* Weff     = (__bf16*)(ws + 6316032);     // 8*512*512*2 = 4 MiB

    ct_softmax_bias<<<BB * CINT, 256, 0, stream>>>(energy, b_value, attn, bias_eff);
    ct_aeff <<<(BB * CINT * CIN) / 256, 256, 0, stream>>>(attn, w_value, A_eff);
    ct_weff <<<(BB * CIN * CIN) / 256, 256, 0, stream>>>(w_re, A_eff, Weff);
    ct_bias2<<<(BB * CIN) / 256, 256, 0, stream>>>(w_re, bias_eff, b_re, bias2);

    dim3 grid(HW / 64, BB);
    ct_main_gemm<<<grid, 512, 0, stream>>>(x, Weff, bias2, gamma, out);
}
